// KBGAT_Model_22617297780845
// MI455X (gfx1250) — matrix operand order repaired and checked by reading, not compiled, not run on hardware
//
#include <hip/hip_runtime.h>
#include <math.h>

// ---------------- problem constants ----------------
#define E_DIRECT 200000
#define E_2HOP   50000
#define E_TOT    250000           // E_DIRECT + E_2HOP
#define N_ENTV   40000
#define N_RELV   500
#define D_IN     128
#define D_EMB    256
#define ETILES   (E_TOT / 16)     // 15625
#define MTILES   (N_ENTV / 16)    // 2500

typedef __bf16 bf16;
typedef bf16  v16bf __attribute__((ext_vector_type(16)));
typedef float v8f   __attribute__((ext_vector_type(8)));

// float atomic max via int/uint trick (works for all finite floats, init -1e30)
__device__ __forceinline__ void atomicMaxF(float* a, float v) {
  if (v >= 0.0f) atomicMax((int*)a, __float_as_int(v));
  else           atomicMin((unsigned int*)a, __float_as_uint(v));
}

// ---------------- generic f32 -> bf16 conversion ----------------
__global__ void k_cvt(const float* __restrict__ src, bf16* __restrict__ dst, int n) {
  int i = blockIdx.x * blockDim.x + threadIdx.x;
  if (i < n) dst[i] = (bf16)src[i];
}

// ---------------- kernel 1: unify edge lists ----------------
__global__ void k_build(const int* __restrict__ edge_index,
                        const int* __restrict__ edge_type,
                        const int* __restrict__ ind2,
                        int* __restrict__ rowA, int* __restrict__ colA,
                        int* __restrict__ t0A,  int* __restrict__ t1A) {
  int e = blockIdx.x * blockDim.x + threadIdx.x;
  if (e >= E_TOT) return;
  if (e < E_DIRECT) {
    rowA[e] = edge_index[E_DIRECT + e];  // ei[0] = edge_index[1]
    colA[e] = edge_index[e];             // ei[1] = edge_index[0]
    t0A[e]  = edge_type[e];
    t1A[e]  = -1;
  } else {
    int j = (e - E_DIRECT) * 4;
    rowA[e] = ind2[j + 3];
    colA[e] = ind2[j + 0];
    t0A[e]  = ind2[j + 1];
    t1A[e]  = ind2[j + 2];
  }
}

// ---------------- kernel 2: init accumulators ----------------
__global__ void k_init(float* __restrict__ xbuf, float* __restrict__ doutx,
                       float* __restrict__ sm1, float* __restrict__ ss1,
                       float* __restrict__ sm2, float* __restrict__ ss2,
                       float* __restrict__ stats) {
  size_t i = (size_t)blockIdx.x * blockDim.x + threadIdx.x;
  if (i < (size_t)N_ENTV * D_EMB) { xbuf[i] = 0.0f; doutx[i] = 0.0f; }
  if (i < (size_t)N_ENTV * 4)     { sm1[i] = -1e30f; ss1[i] = 0.0f; }
  if (i < (size_t)N_ENTV)         { sm2[i] = -1e30f; ss2[i] = 0.0f; }
  if (i < 2 * D_EMB)              { stats[i] = 0.0f; }
}

// ---------------- kernel 3: r = init_rel @ gat_W (exact f32, tiny) ----------
__global__ void k_rgemm(const float* __restrict__ rel,
                        const float* __restrict__ gatW,
                        float* __restrict__ r) {
  int i = blockIdx.x * blockDim.x + threadIdx.x;
  if (i >= N_RELV * D_EMB) return;
  int ri = i / D_EMB, n = i % D_EMB;
  float s = 0.0f;
  for (int k = 0; k < D_IN; ++k) s += rel[ri * D_IN + k] * gatW[k * D_EMB + n];
  r[i] = s;
}

// ---------------- kernel 4: phase-1 WMMA GEMM (K = 384) ---------------------
__global__ __launch_bounds__(256) void k_gemm1(
    const bf16* __restrict__ embedBf, const float* __restrict__ rel,
    const bf16* __restrict__ WhBf,
    const int* __restrict__ rowA, const int* __restrict__ colA,
    const int* __restrict__ t0A,  const int* __restrict__ t1A,
    bf16* __restrict__ eh) {
  const int lane  = threadIdx.x & 31;
  const int etile = blockIdx.x * 8 + (threadIdx.x >> 5);
  if (etile >= ETILES) return;
  const int n0 = blockIdx.y * 16;        // output column tile (0..255)
  const int h  = n0 >> 6;                // head
  const int j0 = n0 & 63;                // column within head

  const int m  = lane & 15;
  const int e  = etile * 16 + m;
  const int kb = (lane >> 4) * 8;
  const bf16*  prow = embedBf + (size_t)rowA[e] * D_IN;
  const bf16*  pcol = embedBf + (size_t)colA[e] * D_IN;
  const float* pr0  = rel     + (size_t)t0A[e] * D_IN;
  const int    t1   = t1A[e];
  const float* pr1  = (t1 >= 0) ? (rel + (size_t)t1 * D_IN) : nullptr;

  v8f acc = {};
  for (int K0 = 0; K0 < 384; K0 += 32) {
    v16bf af;
#pragma unroll
    for (int v = 0; v < 8; ++v) {
      int k = K0 + kb + (v < 4 ? 2 * v : 16 + 2 * (v - 4));
      if (k < 128)      { af[2 * v] = prow[k];       af[2 * v + 1] = prow[k + 1]; }
      else if (k < 256) { af[2 * v] = pcol[k - 128]; af[2 * v + 1] = pcol[k - 127]; }
      else {
        int kk = k - 256;
        float x0 = pr0[kk], x1 = pr0[kk + 1];
        if (pr1) { x0 += pr1[kk]; x1 += pr1[kk + 1]; }
        af[2 * v] = (bf16)x0; af[2 * v + 1] = (bf16)x1;
      }
    }
    // B fragment: lane holds column j0+m; element s holds K in table order (same as A)
    v16bf bfv;
#pragma unroll
    for (int s = 0; s < 16; ++s)
      bfv[s] = WhBf[((size_t)h * 384 + (K0 + kb + (s < 8 ? s : 8 + s))) * 64 + j0 + m];
    acc = __builtin_amdgcn_wmma_f32_16x16x32_bf16(false, af, false, bfv,
                                                  (short)0, acc, false, false);
  }
  const int nOut = n0 + (lane & 15);
  const int mofs = (lane >> 4) * 8;
#pragma unroll
  for (int v = 0; v < 8; ++v) {
    int eo = etile * 16 + mofs + v;
    eh[(size_t)eo * D_EMB + nOut] = (bf16)acc[v];
  }
}

// ---------------- kernel 5..7: phase-1 segment softmax + aggregate ----------
__global__ void k_logit1(const bf16* __restrict__ eh, const float* __restrict__ aH,
                         const int* __restrict__ rowA,
                         float* __restrict__ logitB, float* __restrict__ sm1) {
  int i = blockIdx.x * blockDim.x + threadIdx.x;
  if (i >= E_TOT * 4) return;
  int e = i >> 2, h = i & 3;
  const bf16*  p = eh + (size_t)e * D_EMB + h * 64;
  const float* a = aH + h * 64;
  float d = 0.0f;
  for (int j = 0; j < 64; ++j) d += (float)p[j] * a[j];
  float l = (d > 0.0f) ? d : 0.2f * d;   // leaky_relu
  logitB[i] = l;
  atomicMaxF(&sm1[rowA[e] * 4 + h], l);
}

__global__ void k_exp1(const int* __restrict__ rowA, float* __restrict__ logitB,
                       const float* __restrict__ sm1, float* __restrict__ ss1) {
  int i = blockIdx.x * blockDim.x + threadIdx.x;
  if (i >= E_TOT * 4) return;
  int e = i >> 2, h = i & 3;
  float ev = __expf(logitB[i] - sm1[rowA[e] * 4 + h]);
  logitB[i] = ev;
  atomicAdd(&ss1[rowA[e] * 4 + h], ev);
}

__global__ void k_agg1(const bf16* __restrict__ eh, const float* __restrict__ evB,
                       const float* __restrict__ ss1, const int* __restrict__ rowA,
                       float* __restrict__ xbuf) {
  int e = blockIdx.x;          // one block per edge
  int n = threadIdx.x;         // 0..255
  int h = n >> 6;
  int r = rowA[e];
  float alpha = evB[e * 4 + h] / (ss1[r * 4 + h] + 1e-16f);
  atomicAdd(&xbuf[(size_t)r * D_EMB + n], alpha * (float)eh[(size_t)e * D_EMB + n]);
}

__global__ void k_elu(float* __restrict__ xbuf, bf16* __restrict__ xbf) {
  size_t i = (size_t)blockIdx.x * blockDim.x + threadIdx.x;
  if (i >= (size_t)N_ENTV * D_EMB) return;
  float v = xbuf[i];
  v = (v > 0.0f) ? v : (__expf(v) - 1.0f);
  xbf[i] = (bf16)v;            // bf16 copy for the phase-2 gathers
}

// ---------------- kernel 8: phase-2 WMMA GEMM (K = 768) ---------------------
__global__ __launch_bounds__(256) void k_gemm2(
    const bf16* __restrict__ xbf, const float* __restrict__ r,
    const bf16* __restrict__ outWBf,
    const int* __restrict__ rowA, const int* __restrict__ colA,
    const int* __restrict__ t0A,  const int* __restrict__ t1A,
    bf16* __restrict__ eh) {
  const int lane  = threadIdx.x & 31;
  const int etile = blockIdx.x * 8 + (threadIdx.x >> 5);
  if (etile >= ETILES) return;
  const int n0 = blockIdx.y * 16;

  const int m  = lane & 15;
  const int e  = etile * 16 + m;
  const int kb = (lane >> 4) * 8;
  const bf16*  prow = xbf + (size_t)rowA[e] * D_EMB;
  const bf16*  pcol = xbf + (size_t)colA[e] * D_EMB;
  const float* pr0  = r   + (size_t)t0A[e] * D_EMB;
  const int    t1   = t1A[e];
  const float* pr1  = (t1 >= 0) ? (r + (size_t)t1 * D_EMB) : nullptr;

  v8f acc = {};
  for (int K0 = 0; K0 < 768; K0 += 32) {
    v16bf af;
#pragma unroll
    for (int v = 0; v < 8; ++v) {
      int k = K0 + kb + (v < 4 ? 2 * v : 16 + 2 * (v - 4));
      if (k < 256)      { af[2 * v] = prow[k];       af[2 * v + 1] = prow[k + 1]; }
      else if (k < 512) { af[2 * v] = pcol[k - 256]; af[2 * v + 1] = pcol[k - 255]; }
      else {
        int kk = k - 512;
        float x0 = pr0[kk], x1 = pr0[kk + 1];
        if (pr1) { x0 += pr1[kk]; x1 += pr1[kk + 1]; }
        af[2 * v] = (bf16)x0; af[2 * v + 1] = (bf16)x1;
      }
    }
    v16bf bfv;
#pragma unroll
    for (int s = 0; s < 16; ++s)
      bfv[s] = outWBf[(size_t)(K0 + kb + (s < 8 ? s : 8 + s)) * D_EMB + n0 + m];
    acc = __builtin_amdgcn_wmma_f32_16x16x32_bf16(false, af, false, bfv,
                                                  (short)0, acc, false, false);
  }
  const int nOut = n0 + (lane & 15);
  const int mofs = (lane >> 4) * 8;
#pragma unroll
  for (int v = 0; v < 8; ++v) {
    int eo = etile * 16 + mofs + v;
    eh[(size_t)eo * D_EMB + nOut] = (bf16)acc[v];
  }
}

// ---------------- kernel 9..11: phase-2 softmax + aggregate -----------------
__global__ void k_logit2(const bf16* __restrict__ eh, const float* __restrict__ outA,
                         const int* __restrict__ rowA,
                         float* __restrict__ logitB, float* __restrict__ sm2) {
  int e = blockIdx.x * blockDim.x + threadIdx.x;
  if (e >= E_TOT) return;
  const bf16* p = eh + (size_t)e * D_EMB;
  float d = 0.0f;
  for (int j = 0; j < D_EMB; ++j) d += (float)p[j] * outA[j];
  float l = (d > 0.0f) ? d : 0.2f * d;
  logitB[e] = l;
  atomicMaxF(&sm2[rowA[e]], l);
}

__global__ void k_exp2(const int* __restrict__ rowA, float* __restrict__ logitB,
                       const float* __restrict__ sm2, float* __restrict__ ss2) {
  int e = blockIdx.x * blockDim.x + threadIdx.x;
  if (e >= E_TOT) return;
  float ev = __expf(logitB[e] - sm2[rowA[e]]);
  logitB[e] = ev;
  atomicAdd(&ss2[rowA[e]], ev);
}

__global__ void k_agg2(const bf16* __restrict__ eh, const float* __restrict__ evB,
                       const float* __restrict__ ss2, const int* __restrict__ rowA,
                       float* __restrict__ doutx) {
  int e = blockIdx.x;
  int n = threadIdx.x;
  int r = rowA[e];
  float alpha = evB[e] / (ss2[r] + 1e-16f);
  atomicAdd(&doutx[(size_t)r * D_EMB + n], alpha * (float)eh[(size_t)e * D_EMB + n]);
}

// ---------------- kernel 12: entity WMMA GEMM + elu-add + BN stats ----------
__global__ __launch_bounds__(256) void k_entity(
    const bf16* __restrict__ embedBf, const bf16* __restrict__ WentsBf,
    float* __restrict__ doutx, float* __restrict__ stats) {
  const int lane  = threadIdx.x & 31;
  const int mtile = blockIdx.x * 8 + (threadIdx.x >> 5);
  if (mtile >= MTILES) return;
  const int n0 = blockIdx.y * 16;

  const int m  = lane & 15;
  const int kb = (lane >> 4) * 8;
  const bf16* prow = embedBf + (size_t)(mtile * 16 + m) * D_IN;

  v8f acc = {};
  for (int K0 = 0; K0 < D_IN; K0 += 32) {
    v16bf af;
#pragma unroll
    for (int v = 0; v < 8; ++v) {
      int k = K0 + kb + (v < 4 ? 2 * v : 16 + 2 * (v - 4));
      af[2 * v] = prow[k]; af[2 * v + 1] = prow[k + 1];
    }
    v16bf bfv;
#pragma unroll
    for (int s = 0; s < 16; ++s)
      bfv[s] = WentsBf[(size_t)(K0 + kb + (s < 8 ? s : 8 + s)) * D_EMB + n0 + m];
    acc = __builtin_amdgcn_wmma_f32_16x16x32_bf16(false, af, false, bfv,
                                                  (short)0, acc, false, false);
  }
  const int nOut = n0 + (lane & 15);
  const int mofs = (lane >> 4) * 8;
  float lsum = 0.0f, lsq = 0.0f;
#pragma unroll
  for (int v = 0; v < 8; ++v) {
    size_t idx = (size_t)(mtile * 16 + mofs + v) * D_EMB + nOut;
    float g = doutx[idx];
    float val = ((g > 0.0f) ? g : (__expf(g) - 1.0f)) + acc[v];
    doutx[idx] = val;
    lsum += val; lsq += val * val;
  }
  atomicAdd(&stats[nOut], lsum);
  atomicAdd(&stats[D_EMB + nOut], lsq);
}

// ---------------- kernel 13: batch-norm normalize ---------------------------
__global__ void k_bn(float* __restrict__ doutx, const float* __restrict__ stats,
                     const float* __restrict__ gamma, const float* __restrict__ beta) {
  size_t i = (size_t)blockIdx.x * D_EMB + threadIdx.x;
  int n = threadIdx.x;
  float mu  = stats[n] * (1.0f / N_ENTV);
  float var = stats[D_EMB + n] * (1.0f / N_ENTV) - mu * mu;
  float v = doutx[i];
  doutx[i] = (v - mu) * rsqrtf(var + 1e-5f) * gamma[n] + beta[n];
}

// ---------------- launcher --------------------------------------------------
extern "C" void kernel_launch(void* const* d_in, const int* in_sizes, int n_in,
                              void* d_out, int out_size, void* d_ws, size_t ws_size,
                              hipStream_t stream) {
  const int*   edge_index = (const int*)d_in[0];
  const int*   edge_type  = (const int*)d_in[1];
  const int*   ind2       = (const int*)d_in[2];
  const float* embed      = (const float*)d_in[3];
  const float* rel        = (const float*)d_in[4];
  const float* Wh         = (const float*)d_in[5];
  const float* aH         = (const float*)d_in[6];
  const float* gatW       = (const float*)d_in[7];
  const float* outW       = (const float*)d_in[8];
  const float* outA       = (const float*)d_in[9];
  const float* Wents      = (const float*)d_in[10];
  const float* gamma      = (const float*)d_in[11];
  const float* beta       = (const float*)d_in[12];

  float* doutx = (float*)d_out;                        // (40000, 256)
  float* doutr = doutx + (size_t)N_ENTV * D_EMB;       // (500, 256)

  // workspace carve (aligned to 256B); total ~211 MB
  char* ws = (char*)d_ws;
  size_t off = 0;
  auto carve = [&](size_t bytes) {
    char* p = ws + off;
    off = (off + bytes + 255) & ~(size_t)255;
    return p;
  };
  int*   rowA    = (int*)  carve(sizeof(int)   * E_TOT);
  int*   colA    = (int*)  carve(sizeof(int)   * E_TOT);
  int*   t0A     = (int*)  carve(sizeof(int)   * E_TOT);
  int*   t1A     = (int*)  carve(sizeof(int)   * E_TOT);
  float* logitB  = (float*)carve(sizeof(float) * E_TOT * 4);
  float* sm1     = (float*)carve(sizeof(float) * N_ENTV * 4);
  float* ss1     = (float*)carve(sizeof(float) * N_ENTV * 4);
  float* sm2     = (float*)carve(sizeof(float) * N_ENTV);
  float* ss2     = (float*)carve(sizeof(float) * N_ENTV);
  float* stats   = (float*)carve(sizeof(float) * 2 * D_EMB);
  float* xbuf    = (float*)carve(sizeof(float) * (size_t)N_ENTV * D_EMB);
  bf16*  embedBf = (bf16*) carve(sizeof(bf16)  * (size_t)N_ENTV * D_IN);
  bf16*  xbf     = (bf16*) carve(sizeof(bf16)  * (size_t)N_ENTV * D_EMB);
  bf16*  WhBf    = (bf16*) carve(sizeof(bf16)  * 4 * 384 * 64);
  bf16*  outWBf  = (bf16*) carve(sizeof(bf16)  * 768 * D_EMB);
  bf16*  WentsBf = (bf16*) carve(sizeof(bf16)  * D_IN * D_EMB);
  bf16*  ehBuf   = (bf16*) carve(sizeof(bf16)  * (size_t)E_TOT * D_EMB);
  (void)ws_size; (void)in_sizes; (void)n_in; (void)out_size;

  const int B = 256;
  auto cvt = [&](const float* s, bf16* d, int n) {
    k_cvt<<<(n + B - 1) / B, B, 0, stream>>>(s, d, n);
  };

  // 1. edges + init
  k_build<<<(E_TOT + B - 1) / B, B, 0, stream>>>(edge_index, edge_type, ind2,
                                                 rowA, colA, t0A, t1A);
  k_init<<<(N_ENTV * D_EMB) / B, B, 0, stream>>>(xbuf, doutx, sm1, ss1, sm2, ss2, stats);
  // 2. bf16 staging of hot operands
  cvt(embed, embedBf, N_ENTV * D_IN);
  cvt(Wh,    WhBf,    4 * 384 * 64);
  cvt(outW,  outWBf,  768 * D_EMB);
  cvt(Wents, WentsBf, D_IN * D_EMB);
  // 3. r = init_rel @ gat_W (exact f32, direct output)
  k_rgemm<<<(N_RELV * D_EMB + B - 1) / B, B, 0, stream>>>(rel, gatW, doutr);
  // 4. phase-1 WMMA GEMM
  k_gemm1<<<dim3((ETILES + 7) / 8, D_EMB / 16), B, 0, stream>>>(
      embedBf, rel, WhBf, rowA, colA, t0A, t1A, ehBuf);
  // 5..7. phase-1 softmax + aggregate
  k_logit1<<<(E_TOT * 4 + B - 1) / B, B, 0, stream>>>(ehBuf, aH, rowA, logitB, sm1);
  k_exp1  <<<(E_TOT * 4 + B - 1) / B, B, 0, stream>>>(rowA, logitB, sm1, ss1);
  k_agg1  <<<E_TOT, D_EMB, 0, stream>>>(ehBuf, logitB, ss1, rowA, xbuf);
  k_elu   <<<(N_ENTV * D_EMB) / B, B, 0, stream>>>(xbuf, xbf);
  // 8. phase-2 WMMA GEMM
  k_gemm2<<<dim3((ETILES + 7) / 8, D_EMB / 16), B, 0, stream>>>(
      xbf, doutr, outWBf, rowA, colA, t0A, t1A, ehBuf);
  // 9..11. phase-2 softmax + aggregate (into d_out x region, zeroed by k_init)
  k_logit2<<<(E_TOT + B - 1) / B, B, 0, stream>>>(ehBuf, outA, rowA, logitB, sm2);
  k_exp2  <<<(E_TOT + B - 1) / B, B, 0, stream>>>(rowA, logitB, sm2, ss2);
  k_agg2  <<<E_TOT, D_EMB, 0, stream>>>(ehBuf, logitB, ss2, rowA, doutx);
  // 12. entity WMMA GEMM + elu-add + BN stats
  k_entity<<<dim3((MTILES + 7) / 8, D_EMB / 16), B, 0, stream>>>(embedBf, WentsBf,
                                                                 doutx, stats);
  // 13. batch-norm
  k_bn<<<N_ENTV, D_EMB, 0, stream>>>(doutx, stats, gamma, beta);
}
